// EdgeModel_13984413516157
// MI455X (gfx1250) — matrix operand order repaired and checked by reading, not compiled, not run on hardware
//
#include <hip/hip_runtime.h>
#include <hip/hip_bf16.h>

typedef __attribute__((ext_vector_type(16))) __bf16 v16bf;
typedef __attribute__((ext_vector_type(8)))  __bf16 v8bf;
typedef __attribute__((ext_vector_type(8)))  float  v8f;
typedef __attribute__((ext_vector_type(4)))  float  v4f;
typedef __attribute__((ext_vector_type(4)))  unsigned int v4u;
typedef __attribute__((ext_vector_type(8)))  int v8i;
typedef __attribute__((ext_vector_type(4)))  int v4i;

#define N_EDGES   500000
#define NODE_F    128
#define HIDDEN    256
#define OUT_DIM   128
// GEMM1: K = 384 (12 k-steps of 32), N = 256 (16 n-tiles) -> 16KB W1 slice / k-step
// GEMM2: K = 256 (8 k-steps of 32),  N = 128 (8 n-tiles)  ->  8KB W2 slice / k-step

// ---------------------------------------------------------------------------
// Weight packing (bf16 fragment layout, k-slice-contiguous so the TDM can DMA
// one k-step as a single 1-D tile).
//   packed[((kt*NT + nt)*32 + lane)*16 + i] =
//       W[K = kt*32 + (lane/16)*8 + i + (i>=8 ? 8 : 0)][N = nt*16 + lane%16]
// ---------------------------------------------------------------------------
__global__ void pack_w1_kernel(const float* __restrict__ W1, __bf16* __restrict__ out) {
    int f    = blockIdx.x;          // 0..191  (kt*16 + nt)
    int lane = threadIdx.x;         // 0..31
    int kt = f >> 4, nt = f & 15;
    int col = nt * 16 + (lane & 15);
    int kb  = kt * 32 + (lane >> 4) * 8;
    #pragma unroll
    for (int i = 0; i < 16; ++i) {
        out[(f * 32 + lane) * 16 + i] = (__bf16)W1[(size_t)(kb + i + (i >= 8 ? 8 : 0)) * HIDDEN + col];
    }
}

__global__ void pack_w2_kernel(const float* __restrict__ W2, __bf16* __restrict__ out) {
    int f    = blockIdx.x;          // 0..63  (kt*8 + nt)
    int lane = threadIdx.x;
    int kt = f >> 3, nt = f & 7;
    int col = nt * 16 + (lane & 15);
    int kb  = kt * 32 + (lane >> 4) * 8;
    #pragma unroll
    for (int i = 0; i < 16; ++i) {
        out[(f * 32 + lane) * 16 + i] = (__bf16)W2[(size_t)(kb + i + (i >= 8 ? 8 : 0)) * OUT_DIM + col];
    }
}

// ---------------------------------------------------------------------------
// TDM: 1-D DMA of n8 x 8-byte elements from global to LDS (ISA Ch8 D# layout).
// group0: count=1 | lds_addr | global_addr | type=2 ("image")
// group1: data_size=8B, tensor_dim0=n8, tensor_dim1=1, tile_dim0=n8, tile_dim1=1
// 6-arg builtin form (clang-23 / therock-10.0 headers).
// ---------------------------------------------------------------------------
__device__ __forceinline__ void tdm_load_1d(unsigned lds_addr,
                                            unsigned long long gaddr, int n8) {
    v4u g0;
    g0[0] = 1u;                                            // count=1, user mode
    g0[1] = lds_addr;                                      // LDS byte address
    g0[2] = (unsigned)(gaddr & 0xFFFFFFFFu);               // global_addr[31:0]
    g0[3] = (unsigned)((gaddr >> 32) & 0x01FFFFFFu) | 0x80000000u; // [56:32] | type=2
    v8i g1;
    g1[0] = (int)(3u << 16);                               // data_size = 8 bytes
    g1[1] = (int)(((unsigned)n8 & 0xFFFFu) << 16);         // tensor_dim0[15:0]
    g1[2] = (int)((((unsigned)n8 >> 16) & 0xFFFFu) | (1u << 16)); // dim0 hi | tensor_dim1=1
    g1[3] = (int)(((unsigned)n8 & 0xFFFFu) << 16);         // tile_dim0 = n8
    g1[4] = 1;                                             // tile_dim1 = 1
    g1[5] = n8;                                            // tensor_dim0_stride
    g1[6] = 0;
    g1[7] = 0;
    v4i z4 = (v4i){0, 0, 0, 0};
    v8i z8 = (v8i){0, 0, 0, 0, 0, 0, 0, 0};
    __builtin_amdgcn_tensor_load_to_lds(g0, g1, z4, z4, z8, 0);
}

__device__ __forceinline__ unsigned lds_offset_of(const void* p) {
    return (unsigned)(unsigned long long)p;   // LDS aperture: addr[31:0] = LDS offset
}

// Convert a 16x32 A-tile slice for one lane: two 8-float chunks -> v16bf.
__device__ __forceinline__ v16bf cvt_a_frag(const float* __restrict__ p) {
    v4f x0 = *(const v4f*)(p);
    v4f x1 = *(const v4f*)(p + 4);
    v4f y0 = *(const v4f*)(p + 16);
    v4f y1 = *(const v4f*)(p + 20);
    v16bf a;
    a[0]  = (__bf16)x0[0]; a[1]  = (__bf16)x0[1]; a[2]  = (__bf16)x0[2]; a[3]  = (__bf16)x0[3];
    a[4]  = (__bf16)x1[0]; a[5]  = (__bf16)x1[1]; a[6]  = (__bf16)x1[2]; a[7]  = (__bf16)x1[3];
    a[8]  = (__bf16)y0[0]; a[9]  = (__bf16)y0[1]; a[10] = (__bf16)y0[2]; a[11] = (__bf16)y0[3];
    a[12] = (__bf16)y1[0]; a[13] = (__bf16)y1[1]; a[14] = (__bf16)y1[2]; a[15] = (__bf16)y1[3];
    return a;
}

__device__ __forceinline__ v16bf join8(v8bf lo, v8bf hi) {
    v16bf a;
    #pragma unroll
    for (int i = 0; i < 8; ++i) { a[i] = lo[i]; a[8 + i] = hi[i]; }
    return a;
}

#define WMMA_BF16(A, B, C) \
    __builtin_amdgcn_wmma_f32_16x16x32_bf16(false, (A), false, (B), (short)0, (C), false, false)

// ---------------------------------------------------------------------------
// Fused edge MLP: 128 edges per block, 8 waves in 4(M)x2(N) grid.
// Dynamic LDS: [0,64K) = H (bf16, row-major), [64K,96K) = double-buffered
// weight stage filled by the Tensor Data Mover.
// ---------------------------------------------------------------------------
__global__ __launch_bounds__(256) void edge_mlp_kernel(
        const float* __restrict__ src, const float* __restrict__ dest,
        const float* __restrict__ ea,  const float* __restrict__ b1,
        const float* __restrict__ b2,
        const __bf16* __restrict__ w1p, const __bf16* __restrict__ w2p,
        float* __restrict__ out, int E) {
    extern __shared__ char smem[];
    __bf16* hlds  = (__bf16*)smem;                 // 64 KB
    __bf16* stage = (__bf16*)(smem + 65536);       // 32 KB (2 x 16KB buffers)

    const int lane = threadIdx.x & 31;
    const int wave = threadIdx.x >> 5;
    const int mg   = wave >> 1;               // 0..3  : 32-row group
    const int ng   = wave & 1;                // 0..1  : N half
    const int g    = lane >> 4;               // A/B lane group
    const int lm   = lane & 15;

    const bool full = ((long)(blockIdx.x + 1) * 128) <= (long)E;
    const long rowbase = (long)blockIdx.x * 128 + mg * 32;
    long ra0 = rowbase + lm;
    long ra1 = rowbase + 16 + lm;
    if (!full) {
        const long emax = (long)E - 1;
        if (ra0 > emax) ra0 = emax;
        if (ra1 > emax) ra1 = emax;
    }

    // ---------------- GEMM1: H = relu(X @ W1 + b1) ----------------
    v8f acc[2][8];
    #pragma unroll
    for (int mi = 0; mi < 2; ++mi)
        #pragma unroll
        for (int nt = 0; nt < 8; ++nt)
            acc[mi][nt] = (v8f){0.f, 0.f, 0.f, 0.f, 0.f, 0.f, 0.f, 0.f};

    // TDM prologue: stage W1 k-slice 0 (16KB = 2048 x 8B)
    if (wave == 0) {
        tdm_load_1d(lds_offset_of(stage), (unsigned long long)w1p, 2048);
    }

    for (int kt = 0; kt < 12; ++kt) {
        const int cur = kt & 1;
        if (wave == 0) __builtin_amdgcn_s_wait_tensorcnt(0);  // slice kt landed
        __syncthreads();                                      // all waves done with buf[1-cur]
        if (wave == 0 && kt + 1 < 12) {
            tdm_load_1d(lds_offset_of(stage) + (1 - cur) * 16384,
                        (unsigned long long)w1p + (unsigned long long)(kt + 1) * 16384,
                        2048);
        }

        const float* base;
        int cb;
        if (kt < 4)      { base = src;  cb = kt * 32; }
        else if (kt < 8) { base = dest; cb = (kt - 4) * 32; }
        else             { base = ea;   cb = (kt - 8) * 32; }
        const int co = cb + g * 8;
        v16bf a0 = cvt_a_frag(base + ra0 * NODE_F + co);
        v16bf a1 = cvt_a_frag(base + ra1 * NODE_F + co);

        // B fragments now come from LDS: 2x ds_load_b128 per fragment.
        const v16bf* wrow = (const v16bf*)(stage + cur * 8192) + (size_t)(ng * 8) * 32 + lane;
        #pragma unroll
        for (int q = 0; q < 8; ++q) {
            v16bf b = wrow[q * 32];
            acc[0][q] = WMMA_BF16(a0, b, acc[0][q]);
            acc[1][q] = WMMA_BF16(a1, b, acc[1][q]);
        }
        __syncthreads();                                      // done reading buf[cur]
    }

    // relu(+b1) -> LDS H (bf16, row-major [row][HIDDEN])
    #pragma unroll
    for (int nt = 0; nt < 8; ++nt) {
        const float bias = b1[(ng * 8 + nt) * 16 + lm];
        const int  col   = (ng * 8 + nt) * 16 + lm;
        #pragma unroll
        for (int mi = 0; mi < 2; ++mi) {
            #pragma unroll
            for (int r = 0; r < 8; ++r) {
                float v = acc[mi][nt][r] + bias;
                v = v > 0.f ? v : 0.f;
                int row = mg * 32 + mi * 16 + r + 8 * g;
                hlds[row * HIDDEN + col] = (__bf16)v;
            }
        }
    }

    // TDM prologue for GEMM2: stage W2 k-slice 0 (8KB = 1024 x 8B)
    if (wave == 0) {
        tdm_load_1d(lds_offset_of(stage), (unsigned long long)w2p, 1024);
    }
    __syncthreads();   // H visible to all; stage buf0 owned by TDM

    // ---------------- GEMM2: OUT = H @ W2 + b2 + edge_attr ----------------
    v8f acc2[2][4];
    #pragma unroll
    for (int mi = 0; mi < 2; ++mi)
        #pragma unroll
        for (int nt = 0; nt < 4; ++nt)
            acc2[mi][nt] = (v8f){0.f, 0.f, 0.f, 0.f, 0.f, 0.f, 0.f, 0.f};

    for (int kt = 0; kt < 8; ++kt) {
        const int cur = kt & 1;
        if (wave == 0) __builtin_amdgcn_s_wait_tensorcnt(0);
        __syncthreads();
        if (wave == 0 && kt + 1 < 8) {
            tdm_load_1d(lds_offset_of(stage) + (1 - cur) * 8192,
                        (unsigned long long)w2p + (unsigned long long)(kt + 1) * 8192,
                        1024);
        }

        const int c0 = kt * 32 + g * 8;
        const int r0 = (mg * 32 + lm) * HIDDEN;
        const int r1 = (mg * 32 + 16 + lm) * HIDDEN;
        v16bf a0 = join8(*(const v8bf*)&hlds[r0 + c0], *(const v8bf*)&hlds[r0 + c0 + 16]);
        v16bf a1 = join8(*(const v8bf*)&hlds[r1 + c0], *(const v8bf*)&hlds[r1 + c0 + 16]);

        const v16bf* wrow = (const v16bf*)(stage + cur * 4096) + (size_t)(ng * 4) * 32 + lane;
        #pragma unroll
        for (int q = 0; q < 4; ++q) {
            v16bf b = wrow[q * 32];
            acc2[0][q] = WMMA_BF16(a0, b, acc2[0][q]);
            acc2[1][q] = WMMA_BF16(a1, b, acc2[1][q]);
        }
        __syncthreads();
    }

    // epilogue: + b2 + residual; unguarded fast path for full blocks
    #pragma unroll
    for (int nt = 0; nt < 4; ++nt) {
        const int  col  = (ng * 4 + nt) * 16 + lm;
        const float b2v = b2[col];
        if (full) {
            #pragma unroll
            for (int mi = 0; mi < 2; ++mi) {
                #pragma unroll
                for (int r = 0; r < 8; ++r) {
                    long row = (long)blockIdx.x * 128 + mg * 32 + mi * 16 + r + 8 * g;
                    size_t idx = (size_t)row * OUT_DIM + col;
                    out[idx] = acc2[mi][nt][r] + b2v + ea[idx];
                }
            }
        } else {
            #pragma unroll
            for (int mi = 0; mi < 2; ++mi) {
                #pragma unroll
                for (int r = 0; r < 8; ++r) {
                    long row = (long)blockIdx.x * 128 + mg * 32 + mi * 16 + r + 8 * g;
                    if (row < E) {
                        size_t idx = (size_t)row * OUT_DIM + col;
                        out[idx] = acc2[mi][nt][r] + b2v + ea[idx];
                    }
                }
            }
        }
    }
}

// ---------------------------------------------------------------------------
// Launch
// Inputs: 0:src 1:dest 2:edge_attr 3:u 4:batch 5:W1 6:b1 7:W2 8:b2
// ws: [0, 196608) bytes -> packed W1 bf16 ; [196608, 262144) -> packed W2 bf16
// ---------------------------------------------------------------------------
extern "C" void kernel_launch(void* const* d_in, const int* in_sizes, int n_in,
                              void* d_out, int out_size, void* d_ws, size_t ws_size,
                              hipStream_t stream) {
    const float* src  = (const float*)d_in[0];
    const float* dest = (const float*)d_in[1];
    const float* ea   = (const float*)d_in[2];
    const float* W1   = (const float*)d_in[5];
    const float* b1   = (const float*)d_in[6];
    const float* W2   = (const float*)d_in[7];
    const float* b2   = (const float*)d_in[8];
    float* out = (float*)d_out;

    __bf16* w1p = (__bf16*)d_ws;                          // 384*256 halves
    __bf16* w2p = (__bf16*)((char*)d_ws + 384 * 256 * 2); // 256*128 halves

    pack_w1_kernel<<<192, 32, 0, stream>>>(W1, w1p);
    pack_w2_kernel<<<64, 32, 0, stream>>>(W2, w2p);

    const int E = N_EDGES;
    const int nblocks = (E + 127) / 128;
    const size_t shmem = 96 * 1024;   // 64KB H + 2x16KB TDM stage
    edge_mlp_kernel<<<nblocks, 256, shmem, stream>>>(src, dest, ea, b1, b2, w1p, w2p, out, E);
}
